// EnergyPredictor_v2_79061757985257
// MI455X (gfx1250) — matrix operand order repaired and checked by reading, not compiled, not run on hardware
//
#include <hip/hip_runtime.h>
#include <hip/hip_bf16.h>

#define BB 4
#define NP 2000
#define NNB 64
#define NPART (BB * NP)

typedef __attribute__((ext_vector_type(16))) __bf16 v16bf;
typedef __attribute__((ext_vector_type(8)))  float  v8f;

__device__ __forceinline__ unsigned short f2bf(float x) {
  unsigned int u = __float_as_uint(x);
  unsigned int r = (u + 0x7FFFu + ((u >> 16) & 1u)) >> 16;
  return (unsigned short)r;
}
__device__ __forceinline__ float bf2f(unsigned short h) {
  return __uint_as_float(((unsigned int)h) << 16);
}

// gfx1250 has a native V_TANH_F32 (TRANS32 pipe, co-executes with WMMA).
__device__ __forceinline__ float fast_tanh(float x) {
#if __has_builtin(__builtin_amdgcn_tanhf)
  return __builtin_amdgcn_tanhf(x);
#elif __has_builtin(__builtin_amdgcn_tanh_f32)
  return __builtin_amdgcn_tanh_f32(x);
#else
  return tanhf(x);
#endif
}

union Frag { v16bf v; uint4 q[2]; };

enum GMode { G_TANH, G_MEAN, G_DERIV, G_ACC };

// C[64xNOUT] = A[64xK] @ Bt^T  where Bt is the n-major (N x K) bf16 weight copy in global.
// A is bf16 row-major in LDS (row stride K halves).  One wave per 16x16 tile, WMMA bf16 K=32.
template<int K, int NOUT, GMode MODE>
__device__ __forceinline__ void wgemm(const unsigned short* __restrict__ A,
                                      const unsigned short* __restrict__ Bt,
                                      const float* __restrict__ bias,
                                      const unsigned short* __restrict__ Hact,
                                      unsigned short* __restrict__ outB,
                                      float* __restrict__ outF,
                                      int tid) {
  const int lane = tid & 31;
  const int wid  = tid >> 5;
  const int g    = lane >> 4;   // half-wave group
  const int ln   = lane & 15;
  const int nTiles = 4 * (NOUT / 16);
  for (int t = wid; t < nTiles; t += 8) {
    const int mt = t & 3;
    const int nt = t >> 2;
    const int m  = mt * 16 + ln;   // A row for this lane
    const int nn = nt * 16 + ln;   // B/D column for this lane
    v8f acc = {};
#pragma unroll
    for (int k0 = 0; k0 < K; k0 += 32) {
      Frag fa, fb;
      // A fragment: halves 0..7 -> k = k0+8g+h ; halves 8..15 -> k = k0+16+8g+(h-8)
      fa.q[0] = *(const uint4*)(A + m * K + k0 + 8 * g);
      fa.q[1] = *(const uint4*)(A + m * K + k0 + 16 + 8 * g);
      // B fragment: halves 0..7 -> k = k0+8g+h ; halves 8..15 -> k = k0+16+8g+(h-8)
      const uint4* pb = (const uint4*)(Bt + nn * K + k0 + 8 * g);
      fb.q[0] = pb[0];
      fb.q[1] = pb[2];
      acc = __builtin_amdgcn_wmma_f32_16x16x32_bf16(false, fa.v, false, fb.v,
                                                    (short)0, acc, false, false);
    }
    if constexpr (MODE == G_TANH) {
      const float bv = bias[nn];
#pragma unroll
      for (int r = 0; r < 8; ++r) {
        const int mm = mt * 16 + r + 8 * g;
        outB[mm * NOUT + nn] = f2bf(fast_tanh(acc[r] + bv));
      }
    } else if constexpr (MODE == G_MEAN) {
      float s = 0.f;
#pragma unroll
      for (int r = 0; r < 8; ++r) s += acc[r];
      atomicAdd(&outF[nn], s);          // column partial sums (mean done by caller)
    } else if constexpr (MODE == G_DERIV) {
#pragma unroll
      for (int r = 0; r < 8; ++r) {
        const int mm = mt * 16 + r + 8 * g;
        const float h = bf2f(Hact[mm * NOUT + nn]);
        outB[mm * NOUT + nn] = f2bf(acc[r] * (1.f - h * h));
      }
    } else { // G_ACC
#pragma unroll
      for (int r = 0; r < 8; ++r) {
        const int mm = mt * 16 + r + 8 * g;
        outF[mm * NOUT + nn] += acc[r];
      }
    }
  }
  __syncthreads();
}

// ---- bf16 weight region layout (half-element offsets inside d_ws) ----
// per matrix: row-major copy (rows x cols), then transposed copy (cols x Kpad) K zero-padded to 32
enum : int {
  WR0 = 0,     WT0 = 2048,    // 16x128
  WR1 = 6144,  WT1 = 22528,   // 128x128
  WR2 = 38912, WT2 = 55296,   // 128x128
  WR3 = 71680, WT3 = 79872,   // 128x64
  VR0 = 88064, VT0 = 89088,   // 16x64
  VR1 = 91136, VT1 = 95232,   // 64x64
  VR2 = 99328, VT2 = 103424,  // 64x64
  WH_TOTAL = 107520
};
#define GR_BYTES 288000           // B*N*9 f32 scatter buffer at d_ws offset 0

struct PrepPtrs { const float* w[7]; unsigned short* dst; float* gR; };

__global__ void prep_kernel(PrepPtrs P) {
  int i = blockIdx.x * blockDim.x + threadIdx.x;
  const int rows[7] = {16, 128, 128, 128, 16, 64, 64};
  const int cols[7] = {128, 128, 128, 64, 64, 64, 64};
  const int kpad[7] = {32, 128, 128, 128, 32, 64, 64};
  const int rmof[7] = {WR0, WR1, WR2, WR3, VR0, VR1, VR2};
  if (i < NPART * 9) { P.gR[i] = 0.f; return; }
  i -= NPART * 9;
  if (i < 52224) {                       // row-major bf16 copies
    int cum = 0;
    for (int m = 0; m < 7; ++m) {
      int sz = rows[m] * cols[m];
      if (i < cum + sz) { int j = i - cum; P.dst[rmof[m] + j] = f2bf(P.w[m][j]); return; }
      cum += sz;
    }
    return;
  }
  i -= 52224;
  if (i < 55296) {                       // transposed, K padded to mult of 32
    int cum = 0;
    for (int m = 0; m < 7; ++m) {
      int sz = cols[m] * kpad[m];
      if (i < cum + sz) {
        int j = i - cum;
        int c = j / kpad[m], r = j % kpad[m];
        unsigned short v = (r < rows[m]) ? f2bf(P.w[m][r * cols[m] + c]) : (unsigned short)0;
        P.dst[rmof[m] + rows[m] * cols[m] + j] = v;
        return;
      }
      cum += sz;
    }
  }
}
#define PREP_TOTAL (NPART * 9 + 52224 + 55296)

// ---- LDS byte offsets for the main kernel ----
enum : int {
  O_X0 = 0,        // bf16 [64][32] (K padded)
  O_H1 = 4096,     // bf16 [64][128]
  O_H2 = 20480,
  O_H3 = 36864,
  O_P1 = 53248,    // bf16 [64][64]
  O_P2 = 61440,
  O_GA = 69632,    // bf16 [64][128] grad scratch
  O_GB = 86016,
  O_GP = 102400,   // bf16 [64][64]
  O_GD = 110592,   // f32 [64][16]
  O_DR = 114688,   // f32 [64][3]
  O_UU = 115456,
  O_RR = 116224,   // f32 [64]
  O_RI = 116480,   // f32 [9] (padded)
  O_RJ = 116528,   // f32 [64][9]
  O_NB = 118832,   // int [64]
  O_PF = 119088,   // f32 [64]
  O_PU = 119344,
  O_FV = 119600,
  O_E1 = 119856,   // f32 [128]
  O_E2 = 120368,
  O_E3 = 120880,
  O_A3 = 121392,
  O_A2 = 121904,
  O_A1 = 122416,
  O_GF = 122928,   // f32 [64]
  O_G3 = 123184,   // f32 [128]
  O_G2 = 123696,   // f32 [64]
  O_RD = 123952,   // f32 [64]
  O_EN = 124208,   // f32 [1]
  SMEM_BYTES = 124224
};

struct Ptrs {
  const float* pos; const float* ori; const int* nbr; const float* box;
  const float* nW[4]; const float* nB[4];
  const float* pW[3]; const float* pB[3];
  const float* eW[4]; const float* eB[4];
  const unsigned short* wh;     // bf16 weight region
  float* gR;                    // [B*N*9] scatter accumulator
  float* outF;                  // force  [B*N*3]
  float* outE;                  // energy [B*N]
};

__global__ __launch_bounds__(256) void energy_main(Ptrs P) {
  extern __shared__ unsigned char smem[];
  unsigned short* sX0 = (unsigned short*)(smem + O_X0);
  unsigned short* sH1 = (unsigned short*)(smem + O_H1);
  unsigned short* sH2 = (unsigned short*)(smem + O_H2);
  unsigned short* sH3 = (unsigned short*)(smem + O_H3);
  unsigned short* sP1 = (unsigned short*)(smem + O_P1);
  unsigned short* sP2 = (unsigned short*)(smem + O_P2);
  unsigned short* sGA = (unsigned short*)(smem + O_GA);
  unsigned short* sGB = (unsigned short*)(smem + O_GB);
  unsigned short* sGP = (unsigned short*)(smem + O_GP);
  float* sGd = (float*)(smem + O_GD);
  float* sDr = (float*)(smem + O_DR);
  float* sU  = (float*)(smem + O_UU);
  float* sR  = (float*)(smem + O_RR);
  float* sRi = (float*)(smem + O_RI);
  float* sRj = (float*)(smem + O_RJ);
  int*   sNb = (int*)  (smem + O_NB);
  float* sPF = (float*)(smem + O_PF);
  float* sPU = (float*)(smem + O_PU);
  float* sF  = (float*)(smem + O_FV);
  float* sE1 = (float*)(smem + O_E1);
  float* sE2 = (float*)(smem + O_E2);
  float* sE3 = (float*)(smem + O_E3);
  float* sA3 = (float*)(smem + O_A3);
  float* sA2 = (float*)(smem + O_A2);
  float* sA1 = (float*)(smem + O_A1);
  float* sGF = (float*)(smem + O_GF);
  float* sG3 = (float*)(smem + O_G3);
  float* sG2 = (float*)(smem + O_G2);
  float* sRed = (float*)(smem + O_RD);
  float* sEn  = (float*)(smem + O_EN);

  const int tid = threadIdx.x;
  const int blk = blockIdx.x;          // b*NP + n
  const int b = blk / NP;

  // ---------------- geometry forward: build d[64][16] ----------------
  if (tid < 9) sRi[tid] = P.ori[blk * 9 + tid];
  if (tid == 0) sEn[0] = 0.f;
  __syncthreads();
  if (tid < 64) {
    const int m = tid;
    const int j = P.nbr[blk * NNB + m];
    sNb[m] = j;
    float dr[3];
#pragma unroll
    for (int c = 0; c < 3; ++c) {
      float d0 = P.pos[(b * NP + j) * 3 + c] - P.pos[blk * 3 + c];
      float bx = P.box[c];
      d0 -= bx * rintf(d0 / bx);       // minimum image, round-half-even like jnp.round
      dr[c] = d0; sDr[m * 3 + c] = d0;
    }
    const float R = sqrtf(dr[0]*dr[0] + dr[1]*dr[1] + dr[2]*dr[2]);
    sR[m] = R;
    const float s = R + 1e-8f;
    float u[3]; float Rj[9];
#pragma unroll
    for (int c = 0; c < 3; ++c) { u[c] = dr[c] / s; sU[m * 3 + c] = u[c]; }
#pragma unroll
    for (int k = 0; k < 9; ++k) { Rj[k] = P.ori[(b * NP + j) * 9 + k]; sRj[m * 9 + k] = Rj[k]; }
    float d[16];
    d[0] = R;
#pragma unroll
    for (int c = 0; c < 3; ++c) {
      d[1 + c] = u[0] * sRi[c] + u[1] * sRi[3 + c] + u[2] * sRi[6 + c];
      d[4 + c] = u[0] * Rj[c]  + u[1] * Rj[3 + c]  + u[2] * Rj[6 + c];
    }
#pragma unroll
    for (int c = 0; c < 3; ++c)
#pragma unroll
      for (int e = 0; e < 3; ++e)
        d[7 + 3 * c + e] = sRi[c] * Rj[e] + sRi[3 + c] * Rj[3 + e] + sRi[6 + c] * Rj[6 + e];
#pragma unroll
    for (int k = 0; k < 16; ++k) sX0[m * 32 + k] = f2bf(d[k]);
#pragma unroll
    for (int k = 16; k < 32; ++k) sX0[m * 32 + k] = 0;   // K padding
  }
  if (tid < 64) { sPF[tid] = 0.f; sPU[tid] = 0.f; }
  __syncthreads();

  // ---------------- forward MLPs (WMMA bf16) ----------------
  wgemm<32, 128, G_TANH>(sX0, P.wh + WT0, P.nB[0], nullptr, sH1, nullptr, tid);
  wgemm<128,128, G_TANH>(sH1, P.wh + WT1, P.nB[1], nullptr, sH2, nullptr, tid);
  wgemm<128,128, G_TANH>(sH2, P.wh + WT2, P.nB[2], nullptr, sH3, nullptr, tid);
  wgemm<128, 64, G_MEAN>(sH3, P.wh + WT3, nullptr, nullptr, nullptr, sPF, tid);
  wgemm<32,  64, G_TANH>(sX0, P.wh + VT0, P.pB[0], nullptr, sP1, nullptr, tid);
  wgemm<64,  64, G_TANH>(sP1, P.wh + VT1, P.pB[1], nullptr, sP2, nullptr, tid);
  wgemm<64,  64, G_MEAN>(sP2, P.wh + VT2, nullptr, nullptr, nullptr, sPU, tid);
  if (tid < 64)
    sF[tid] = sPF[tid] * (1.f / 64.f) + P.nB[3][tid] + sPU[tid] * (1.f / 64.f) + P.pB[2][tid];
  __syncthreads();

  // ---------------- energy head forward (VALU; M=1 GEMV) ----------------
  if (tid < 128) {
    float s = P.eB[0][tid];
    for (int k = 0; k < 64; ++k) s += sF[k] * P.eW[0][k * 128 + tid];
    sE1[tid] = fast_tanh(s);
  }
  __syncthreads();
  if (tid < 128) {
    float s = P.eB[1][tid];
    for (int k = 0; k < 128; ++k) s += sE1[k] * P.eW[1][k * 128 + tid];
    sE2[tid] = fast_tanh(s);
  }
  __syncthreads();
  if (tid < 128) {
    float s = P.eB[2][tid];
    for (int k = 0; k < 128; ++k) s += sE2[k] * P.eW[2][k * 128 + tid];
    sE3[tid] = fast_tanh(s);
  }
  __syncthreads();
  if (tid < 128) atomicAdd(&sEn[0], sE3[tid] * P.eW[3][tid]);
  __syncthreads();

  // ---------------- energy head backward ----------------
  if (tid < 128) { float e = sE3[tid]; sA3[tid] = P.eW[3][tid] * (1.f - e * e); }
  __syncthreads();
  if (tid < 128) {
    float s = 0.f;
    for (int k = 0; k < 128; ++k) s += sA3[k] * P.eW[2][tid * 128 + k];
    float e = sE2[tid]; sA2[tid] = s * (1.f - e * e);
  }
  __syncthreads();
  if (tid < 128) {
    float s = 0.f;
    for (int k = 0; k < 128; ++k) s += sA2[k] * P.eW[1][tid * 128 + k];
    float e = sE1[tid]; sA1[tid] = s * (1.f - e * e);
  }
  __syncthreads();
  if (tid < 64) {
    float s = 0.f;
    for (int k = 0; k < 128; ++k) s += sA1[k] * P.eW[0][tid * 128 + k];
    sGF[tid] = s * (1.f / 64.f);     // fold the 1/64 mean into the pooled gradient
  }
  __syncthreads();
  // row-uniform gY/gZ -> last-layer backward collapses to GEMV
  if (tid < 128) {
    float s = 0.f;
    for (int nn = 0; nn < 64; ++nn) s += sGF[nn] * P.nW[3][tid * 64 + nn];
    sG3[tid] = s;
  }
  if (tid < 64) {
    float s = 0.f;
    for (int nn = 0; nn < 64; ++nn) s += sGF[nn] * P.pW[2][tid * 64 + nn];
    sG2[tid] = s;
  }
  __syncthreads();
  // pre-activation grads at top hidden layers
  for (int i = tid; i < 64 * 128; i += 256) {
    float h = bf2f(sH3[i]); sGA[i] = f2bf(sG3[i & 127] * (1.f - h * h));
  }
  for (int i = tid; i < 64 * 64; i += 256) {
    float h = bf2f(sP2[i]); sGP[i] = f2bf(sG2[i & 63] * (1.f - h * h));
  }
  for (int i = tid; i < 64 * 16; i += 256) sGd[i] = 0.f;
  __syncthreads();

  // ---------------- backward MLPs (WMMA: grad @ W^T via row-major bf16 copies) ----------------
  wgemm<128,128, G_DERIV>(sGA, P.wh + WR2, nullptr, sH2, sGB, nullptr, tid);
  wgemm<128,128, G_DERIV>(sGB, P.wh + WR1, nullptr, sH1, sGA, nullptr, tid);
  wgemm<128, 16, G_ACC  >(sGA, P.wh + WR0, nullptr, nullptr, nullptr, sGd, tid);
  wgemm<64,  64, G_DERIV>(sGP, P.wh + VR1, nullptr, sP1, sGB, nullptr, tid);
  wgemm<64,  16, G_ACC  >(sGB, P.wh + VR0, nullptr, nullptr, nullptr, sGd, tid);

  // ---------------- geometry backward ----------------
  float gdr[3] = {0.f, 0.f, 0.f};
  float gRiL[9] = {0.f, 0.f, 0.f, 0.f, 0.f, 0.f, 0.f, 0.f, 0.f};
  if (tid < 64) {
    const int m = tid, j = sNb[m];
    float gd[16];
#pragma unroll
    for (int k = 0; k < 16; ++k) gd[k] = sGd[m * 16 + k];
    float u[3], dr[3], Rj[9];
#pragma unroll
    for (int c = 0; c < 3; ++c) { u[c] = sU[m * 3 + c]; dr[c] = sDr[m * 3 + c]; }
#pragma unroll
    for (int k = 0; k < 9; ++k) Rj[k] = sRj[m * 9 + k];
    const float R = sR[m];
    const float s = R + 1e-8f;
    const float Rsafe = fmaxf(R, 1e-12f);
    float gu[3];
#pragma unroll
    for (int d = 0; d < 3; ++d)
      gu[d] = gd[1] * sRi[d * 3 + 0] + gd[2] * sRi[d * 3 + 1] + gd[3] * sRi[d * 3 + 2]
            + gd[4] * Rj[d * 3 + 0]  + gd[5] * Rj[d * 3 + 1]  + gd[6] * Rj[d * 3 + 2];
    const float gudr = gu[0] * dr[0] + gu[1] * dr[1] + gu[2] * dr[2];
#pragma unroll
    for (int e = 0; e < 3; ++e)
      gdr[e] = gd[0] * dr[e] / Rsafe + gu[e] / s - gudr * dr[e] / (s * s * Rsafe);
    float gRjL[9];
#pragma unroll
    for (int d = 0; d < 3; ++d)
#pragma unroll
      for (int c = 0; c < 3; ++c) {
        gRiL[d * 3 + c] = gd[1 + c] * u[d]
            + gd[7 + 3 * c + 0] * Rj[d * 3 + 0]
            + gd[7 + 3 * c + 1] * Rj[d * 3 + 1]
            + gd[7 + 3 * c + 2] * Rj[d * 3 + 2];
        gRjL[d * 3 + c] = gd[4 + c] * u[d]
            + gd[7 + 0 + c] * sRi[d * 3 + 0]
            + gd[7 + 3 + c] * sRi[d * 3 + 1]
            + gd[7 + 6 + c] * sRi[d * 3 + 2];
      }
#pragma unroll
    for (int k = 0; k < 9; ++k)
      atomicAdd(&P.gR[(b * NP + j) * 9 + k], gRjL[k]);   // neighbor-frame scatter
  }
  // force = -sum_m g_dr ; g_Ri reduced locally, one atomic per component
  for (int c = 0; c < 3; ++c) {
    __syncthreads();
    if (tid < 64) sRed[tid] = gdr[c];
    __syncthreads();
    for (int off = 32; off > 0; off >>= 1) {
      if (tid < off) sRed[tid] += sRed[tid + off];
      __syncthreads();
    }
    if (tid == 0) P.outF[blk * 3 + c] = -sRed[0];
  }
  for (int c = 0; c < 9; ++c) {
    __syncthreads();
    if (tid < 64) sRed[tid] = gRiL[c];
    __syncthreads();
    for (int off = 32; off > 0; off >>= 1) {
      if (tid < off) sRed[tid] += sRed[tid + off];
      __syncthreads();
    }
    if (tid == 0) atomicAdd(&P.gR[blk * 9 + c], sRed[0]);
  }
  if (tid == 0) P.outE[blk] = sEn[0] + P.eB[3][0];
}

__global__ void torque_kernel(const float* __restrict__ gR,
                              const float* __restrict__ ori,
                              float* __restrict__ outT) {
  int i = blockIdx.x * blockDim.x + threadIdx.x;
  if (i >= NPART) return;
  float tg[9], Rm[9];
#pragma unroll
  for (int k = 0; k < 9; ++k) { tg[k] = -gR[i * 9 + k]; Rm[k] = ori[i * 9 + k]; }
  float t0 = 0.f, t1 = 0.f, t2 = 0.f;
#pragma unroll
  for (int col = 0; col < 3; ++col) {
    float a0 = tg[col], a1 = tg[3 + col], a2 = tg[6 + col];
    float b0 = Rm[col], b1 = Rm[3 + col], b2 = Rm[6 + col];
    t0 += a1 * b2 - a2 * b1;
    t1 += a2 * b0 - a0 * b2;
    t2 += a0 * b1 - a1 * b0;
  }
  outT[i * 3 + 0] = t0; outT[i * 3 + 1] = t1; outT[i * 3 + 2] = t2;
}

extern "C" void kernel_launch(void* const* d_in, const int* in_sizes, int n_in,
                              void* d_out, int out_size, void* d_ws, size_t ws_size,
                              hipStream_t stream) {
  (void)in_sizes; (void)n_in; (void)out_size; (void)ws_size;
  float* gR = (float*)d_ws;
  unsigned short* wh = (unsigned short*)((char*)d_ws + GR_BYTES);

  // prep: zero g_R scatter buffer + convert neighbor/prior weights to bf16 (2 layouts)
  PrepPtrs pp;
  pp.w[0] = (const float*)d_in[4];   // nW0 16x128
  pp.w[1] = (const float*)d_in[6];   // nW1 128x128
  pp.w[2] = (const float*)d_in[8];   // nW2 128x128
  pp.w[3] = (const float*)d_in[10];  // nW3 128x64
  pp.w[4] = (const float*)d_in[12];  // pW0 16x64
  pp.w[5] = (const float*)d_in[14];  // pW1 64x64
  pp.w[6] = (const float*)d_in[16];  // pW2 64x64
  pp.dst = wh; pp.gR = gR;
  prep_kernel<<<(PREP_TOTAL + 255) / 256, 256, 0, stream>>>(pp);

  Ptrs P;
  P.pos = (const float*)d_in[0];
  P.ori = (const float*)d_in[1];
  P.nbr = (const int*)d_in[2];
  P.box = (const float*)d_in[3];
  for (int l = 0; l < 4; ++l) { P.nW[l] = (const float*)d_in[4 + 2 * l]; P.nB[l] = (const float*)d_in[5 + 2 * l]; }
  for (int l = 0; l < 3; ++l) { P.pW[l] = (const float*)d_in[12 + 2 * l]; P.pB[l] = (const float*)d_in[13 + 2 * l]; }
  for (int l = 0; l < 4; ++l) { P.eW[l] = (const float*)d_in[18 + 2 * l]; P.eB[l] = (const float*)d_in[19 + 2 * l]; }
  P.wh = wh;
  P.gR = gR;
  P.outF = (float*)d_out;                       // [0, 24000) force
  P.outE = (float*)d_out + 2 * NPART * 3;       // [48000, 56000) energy

  (void)hipFuncSetAttribute(reinterpret_cast<const void*>(energy_main),
                            hipFuncAttributeMaxDynamicSharedMemorySize, SMEM_BYTES);
  energy_main<<<NPART, 256, SMEM_BYTES, stream>>>(P);

  torque_kernel<<<(NPART + 255) / 256, 256, 0, stream>>>(
      gR, (const float*)d_in[1], (float*)d_out + NPART * 3);   // [24000, 48000) torque
}
